// Attention_6012954214454
// MI455X (gfx1250) — matrix operand order repaired and checked by reading, not compiled, not run on hardware
//
#include <hip/hip_runtime.h>
#include <hip/hip_bf16.h>
#include <stdint.h>
#include <stddef.h>

#define L_SEQ  2048
#define DMODEL 4096
#define NH     32
#define NKV    8
#define HD     128
#define BATCH  2
#define NEG_INF_F (-1000000000.0f)

#define USE_ASYNC_LDS 1
#define USE_TDM 1

#if USE_TDM && __has_builtin(__builtin_amdgcn_tensor_load_to_lds)
#define HAVE_TDM 1
#else
#define HAVE_TDM 0
#endif

typedef __attribute__((ext_vector_type(16))) __bf16 bf16x16;
typedef __attribute__((ext_vector_type(8)))  float  f32x8;
typedef unsigned int u32x4 __attribute__((ext_vector_type(4)));
typedef int i32x4 __attribute__((ext_vector_type(4)));
typedef int i32x8 __attribute__((ext_vector_type(8)));

// ---------------- bf16 helpers (RNE, bit-exact) -----------------------------
static __device__ __forceinline__ unsigned short f32_to_bf16_bits(float f) {
  unsigned u = __float_as_uint(f);
  u += 0x7FFFu + ((u >> 16) & 1u);
  return (unsigned short)(u >> 16);
}
static __device__ __forceinline__ __bf16 to_bf16(float f) {
  union { unsigned short s; __bf16 b; } cv;
  cv.s = f32_to_bf16_bits(f);
  return cv.b;
}

// ---------------- async global -> LDS 16B copy (ASYNCcnt path) --------------
static __device__ __forceinline__ void copy16_g2l(const __bf16* g, __bf16* l) {
#if USE_ASYNC_LDS
  unsigned lds_addr = (unsigned)(size_t)l;           // low 32 bits = LDS offset
  unsigned long long ga = (unsigned long long)(size_t)g;
  asm volatile("global_load_async_to_lds_b128 %0, %1, off"
               :: "v"(lds_addr), "v"(ga) : "memory");
#else
  *(uint4*)l = *(const uint4*)g;
#endif
}
static __device__ __forceinline__ void wait_async_lds() {
#if USE_ASYNC_LDS
  asm volatile("s_wait_asynccnt 0" ::: "memory");
#endif
}

// ---------------- Tensor Data Mover: 2D tile global -> LDS (TENSORcnt) ------
#if HAVE_TDM
static __device__ __forceinline__ void tdm_load_2d(unsigned lds_addr, const void* g,
                                                   unsigned tileW, unsigned tileH,
                                                   unsigned long long strideElems) {
  unsigned long long ga = (unsigned long long)(size_t)g;
  u32x4 g0;
  g0.x = 1u;                                                  // count=1, user load
  g0.y = lds_addr;                                            // LDS byte address
  g0.z = (unsigned)ga;                                        // global_addr[31:0]
  g0.w = (unsigned)((ga >> 32) & 0x01FFFFFFull) | (2u << 30); // addr[56:32]|type=2
  const unsigned td0 = 1u << 20, td1 = 1u << 20;              // big dims: no OOB clip
  i32x8 g1;
  g1[0] = (int)(1u << 16);                                    // data_size=1 (2B)
  g1[1] = (int)((td0 & 0xFFFFu) << 16);                       // tensor_dim0 lo
  g1[2] = (int)((td0 >> 16) | ((td1 & 0xFFFFu) << 16));       // dim0 hi | dim1 lo
  g1[3] = (int)((td1 >> 16) | (tileW << 16));                 // dim1 hi | tile_dim0
  g1[4] = (int)tileH;                                         // tile_dim1, tile_dim2=0
  g1[5] = (int)(unsigned)strideElems;                         // dim0_stride[31:0]
  g1[6] = (int)(unsigned)(strideElems >> 32);                 // dim0_stride[47:32]
  g1[7] = 0;
  i32x4 gz4 = {0, 0, 0, 0};
  i32x8 gz8 = {0, 0, 0, 0, 0, 0, 0, 0};
  // amdgpu-toolchain (clang-23) 6-arg form: (g0, g1, g2, g3, g_extra, cpol)
  __builtin_amdgcn_tensor_load_to_lds(g0, g1, gz4, gz4, gz8, 0);
}
#endif

// ---------------- WMMA fragment loaders (wave32; contiguous k = 16*half + i, same in A and B) --------------
// A fragment 16x32: lanes 0-15: M=lane, K=0..15; lanes 16-31: M=lane-16,
// K=16..31. Source row-major, row stride ld.
static __device__ __forceinline__ bf16x16 load_frag_a(const __bf16* p, int ld) {
  int lane = threadIdx.x & 31;
  int m = lane & 15;
  int half = (lane >> 4) & 1;
  const __bf16* row = p + (size_t)m * ld + half * 16;
  bf16x16 a;
#pragma unroll
  for (int i = 0; i < 16; ++i) a[i] = row[i];
  return a;
}

// B fragment 32x16 (KxN) from N-major storage p[n*ldt + k] (k contiguous).
static __device__ __forceinline__ bf16x16 load_frag_b_t(const __bf16* p, int ldt) {
  int lane = threadIdx.x & 31;
  int n = lane & 15;
  int kb = (lane >> 4) << 4;
  const __bf16* q = p + (size_t)n * ldt + kb;
  bf16x16 b;
#pragma unroll
  for (int i = 0; i < 16; ++i) b[i] = q[i];
  return b;
}

// B fragment 32x16 (KxN) from row-major storage p[k*ld + n] (strided walk).
static __device__ __forceinline__ bf16x16 load_frag_b_rm(const __bf16* p, int ld) {
  int lane = threadIdx.x & 31;
  int n = lane & 15;
  int kb = (lane >> 4) << 4;
  const __bf16* q = p + (size_t)kb * ld + n;
  bf16x16 b;
#pragma unroll
  for (int i = 0; i < 16; ++i) b[i] = q[(size_t)i * ld];
  return b;
}

static __device__ __forceinline__ f32x8 wmma_bf16(bf16x16 a, bf16x16 b, f32x8 c) {
  return __builtin_amdgcn_wmma_f32_16x16x32_bf16(false, a, false, b, (short)0, c,
                                                 false, false);
}

// ---------------- f32 -> bf16 convert (grid-stride) -------------------------
__global__ void cvt_bf16(const float* __restrict__ in, unsigned short* __restrict__ out,
                         size_t n) {
  size_t i = (size_t)blockIdx.x * blockDim.x + threadIdx.x;
  size_t stride = (size_t)gridDim.x * blockDim.x;
  for (; i < n; i += stride) out[i] = f32_to_bf16_bits(in[i]);
}

// f32 [K][N] row-major -> bf16 [N][K] (transposed, K contiguous). N = 1<<LOGN.
template <int LOGN>
__global__ void cvt_t_bf16(const float* __restrict__ in, unsigned short* __restrict__ out,
                           int K) {
  const int N = 1 << LOGN;
  size_t total = (size_t)K << LOGN;
  size_t i = (size_t)blockIdx.x * blockDim.x + threadIdx.x;
  size_t stride = (size_t)gridDim.x * blockDim.x;
  for (; i < total; i += stride) {
    int n = (int)(i & (size_t)(N - 1));
    int k = (int)(i >> LOGN);
    out[(size_t)n * K + k] = f32_to_bf16_bits(in[i]);
  }
}

// ---------------- bf16 GEMM: C(MxN) f32 = A(MxK) * B(KxN) -------------------
// A row-major [m][k]; BT pre-transposed [n][k]. Block tile 64x128, 8 waves,
// double-buffered LDS fed by the Tensor Data Mover (TENSORcnt pipelined).
#define GT_M 64
#define GT_N 128
#define GT_K 32
template <int N, int K>
__global__ __launch_bounds__(256) void gemm_bf16(const __bf16* __restrict__ A,
                                                 const __bf16* __restrict__ BT,
                                                 float* __restrict__ C, int M) {
  __shared__ __align__(16) __bf16 sA[2][GT_M * GT_K];   // [m][k], stride 32
  __shared__ __align__(16) __bf16 sBT[2][GT_N * GT_K];  // [n][k], stride 32
  const int tid = threadIdx.x;
  const int wave = tid >> 5;
  const int lane = tid & 31;
  const int wm = wave >> 2;  // 0..1
  const int wn = wave & 3;   // 0..3
  const int blockM = blockIdx.y * GT_M;
  const int blockN = blockIdx.x * GT_N;

  f32x8 acc[2][2];
#pragma unroll
  for (int a = 0; a < 2; ++a)
#pragma unroll
    for (int b = 0; b < 2; ++b)
#pragma unroll
      for (int r = 0; r < 8; ++r) acc[a][b][r] = 0.0f;

#if HAVE_TDM
  const __bf16* gA = A + (size_t)blockM * K;
  const __bf16* gB = BT + (size_t)blockN * K;
  if (wave == 0) {  // prologue: DMA first k-tile into buffer 0
    tdm_load_2d((unsigned)(size_t)&sA[0][0], gA, GT_K, GT_M, K);
    tdm_load_2d((unsigned)(size_t)&sBT[0][0], gB, GT_K, GT_N, K);
  }
  int buf = 0;
  for (int k0 = 0; k0 < K; k0 += GT_K) {
    if (wave == 0) {
      if (k0 + GT_K < K) {  // DMA next tile into alternate buffer, retire previous
        tdm_load_2d((unsigned)(size_t)&sA[buf ^ 1][0], gA + k0 + GT_K, GT_K, GT_M, K);
        tdm_load_2d((unsigned)(size_t)&sBT[buf ^ 1][0], gB + k0 + GT_K, GT_K, GT_N, K);
        __builtin_amdgcn_s_wait_tensorcnt(2);
      } else {
        __builtin_amdgcn_s_wait_tensorcnt(0);
      }
    }
    __syncthreads();
#else
  const int eA = tid * 8;                                   // 0..2047
  const __bf16* gA = A + (size_t)(blockM + (eA >> 5)) * K + (eA & 31);
  const int eB0 = tid * 8;
  const int eB1 = tid * 8 + 2048;
  const __bf16* gB0 = BT + (size_t)(blockN + (eB0 >> 5)) * K + (eB0 & 31);
  const __bf16* gB1 = BT + (size_t)(blockN + (eB1 >> 5)) * K + (eB1 & 31);
  const int buf = 0;
  for (int k0 = 0; k0 < K; k0 += GT_K) {
    copy16_g2l(gA + k0, &sA[0][eA]);
    copy16_g2l(gB0 + k0, &sBT[0][eB0]);
    copy16_g2l(gB1 + k0, &sBT[0][eB1]);
    wait_async_lds();
    __syncthreads();
#endif
    bf16x16 af[2], bfr[2];
#pragma unroll
    for (int mi = 0; mi < 2; ++mi)
      af[mi] = load_frag_a(&sA[buf][(wm * 32 + mi * 16) * GT_K], GT_K);
#pragma unroll
    for (int ni = 0; ni < 2; ++ni)
      bfr[ni] = load_frag_b_t(&sBT[buf][(wn * 32 + ni * 16) * GT_K], GT_K);
#pragma unroll
    for (int mi = 0; mi < 2; ++mi)
#pragma unroll
      for (int ni = 0; ni < 2; ++ni)
        acc[mi][ni] = wmma_bf16(af[mi], bfr[ni], acc[mi][ni]);
    __syncthreads();
#if HAVE_TDM
    buf ^= 1;
#endif
  }

  // Epilogue: lane l, reg r: M = r + 8*(l>=16), N = l%16; N is compile-time.
  const int mh = (lane >> 4) << 3;
  const int n = lane & 15;
  float* cp = C + (size_t)(blockM + wm * 32 + mh) * N + blockN + wn * 32 + n;
#pragma unroll
  for (int mi = 0; mi < 2; ++mi)
#pragma unroll
    for (int ni = 0; ni < 2; ++ni)
#pragma unroll
      for (int r = 0; r < 8; ++r)
        cp[(size_t)(mi * 16 + r) * N + ni * 16] = acc[mi][ni][r];
}

// ---------------- RoPE kernels ---------------------------------------------
__global__ void rope_q_kernel(const float* __restrict__ qf, __bf16* __restrict__ qb) {
  size_t idx = (size_t)blockIdx.x * blockDim.x + threadIdx.x;
  if (idx >= (size_t)BATCH * L_SEQ * NH * 64) return;
  int i = (int)(idx & 63); idx >>= 6;
  int h = (int)(idx & 31); idx >>= 5;
  int l = (int)(idx & 2047); idx >>= 11;
  int b = (int)idx;
  const float scale = 0.08838834764831845f;  // 128^-0.5
  float freq = __powf(10000.0f, -(float)i / 64.0f);
  float ang = (float)l * freq;
  float s, c;
  __sincosf(ang, &s, &c);
  const float* src = qf + (size_t)(b * L_SEQ + l) * DMODEL + h * HD + 2 * i;
  float x1 = src[0], x2 = src[1];
  float e0 = (x1 * c + x2 * s) * scale;
  float e1 = (x1 * s - x2 * c) * scale;
  __bf16* dst = qb + ((size_t)(b * NH + h) * L_SEQ + l) * HD + 2 * i;
  dst[0] = to_bf16(e0);
  dst[1] = to_bf16(e1);
}

__global__ void rope_k_kernel(const float* __restrict__ kf, __bf16* __restrict__ kb,
                              float* __restrict__ outk) {
  size_t idx = (size_t)blockIdx.x * blockDim.x + threadIdx.x;
  if (idx >= (size_t)BATCH * L_SEQ * NKV * 64) return;
  int i = (int)(idx & 63); idx >>= 6;
  int hk = (int)(idx & 7); idx >>= 3;
  int l = (int)(idx & 2047); idx >>= 11;
  int b = (int)idx;
  float freq = __powf(10000.0f, -(float)i / 64.0f);
  float ang = (float)l * freq;
  float s, c;
  __sincosf(ang, &s, &c);
  const float* src = kf + (size_t)(b * L_SEQ + l) * (NKV * HD) + hk * HD + 2 * i;
  float x1 = src[0], x2 = src[1];
  float e0 = x1 * c + x2 * s;
  float e1 = x1 * s - x2 * c;
  __bf16* dst = kb + ((size_t)(b * NKV + hk) * L_SEQ + l) * HD + 2 * i;
  dst[0] = to_bf16(e0);
  dst[1] = to_bf16(e1);
#pragma unroll
  for (int r = 0; r < 4; ++r) {
    int h = hk * 4 + r;
    float* d = outk + ((size_t)(b * NH + h) * L_SEQ + l) * HD + 2 * i;
    d[0] = e0;
    d[1] = e1;
  }
}

__global__ void copy_v_kernel(const float* __restrict__ vf, __bf16* __restrict__ vb,
                              float* __restrict__ outv) {
  size_t idx = (size_t)blockIdx.x * blockDim.x + threadIdx.x;
  if (idx >= (size_t)BATCH * L_SEQ * NKV * HD) return;
  int d = (int)(idx & 127); idx >>= 7;
  int hk = (int)(idx & 7); idx >>= 3;
  int l = (int)(idx & 2047); idx >>= 11;
  int b = (int)idx;
  float v = vf[(size_t)(b * L_SEQ + l) * (NKV * HD) + hk * HD + d];
  vb[((size_t)(b * NKV + hk) * L_SEQ + l) * HD + d] = to_bf16(v);
#pragma unroll
  for (int r = 0; r < 4; ++r)
    outv[((size_t)(b * NH + hk * 4 + r) * L_SEQ + l) * HD + d] = v;
}

// ---------------- Flash attention (causal, online softmax) ------------------
// 4 waves / 128-thread block. Uniform trip count across waves so K/V tiles are
// staged ONCE per block into shared LDS via async global->LDS; fully-masked
// tiles are numerically inert (exp(-1e9 - m) == 0).
__global__ __launch_bounds__(128) void flash_attn(const __bf16* __restrict__ qb,
                                                  const __bf16* __restrict__ kb,
                                                  const __bf16* __restrict__ vb,
                                                  __bf16* __restrict__ attnb) {
  __shared__ __align__(16) __bf16 sK[32 * HD];     // [key][d], shared by block
  __shared__ __align__(16) __bf16 sV[32 * HD];     // [key][d], shared by block
  __shared__ __align__(16) __bf16 sP[4][16 * 32];  // per-wave P tile [m][key]
  const int tid = threadIdx.x;
  const int wave = tid >> 5;
  const int lane = tid & 31;
  const int h = blockIdx.y;
  const int b = blockIdx.z;
  const int q0 = blockIdx.x * 64 + wave * 16;
  const int hk = h >> 2;

  const __bf16* Q = qb + ((size_t)(b * NH + h) * L_SEQ + q0) * HD;
  const __bf16* Kp = kb + (size_t)(b * NKV + hk) * L_SEQ * HD;
  const __bf16* Vp = vb + (size_t)(b * NKV + hk) * L_SEQ * HD;

  bf16x16 qfrag[4];
#pragma unroll
  for (int kk = 0; kk < 4; ++kk) qfrag[kk] = load_frag_a(Q + kk * 32, HD);

  f32x8 o[8];
  float mrow[8], lrow[8];
#pragma unroll
  for (int d = 0; d < 8; ++d)
#pragma unroll
    for (int r = 0; r < 8; ++r) o[d][r] = 0.0f;
#pragma unroll
  for (int r = 0; r < 8; ++r) { mrow[r] = -3.0e38f; lrow[r] = 0.0f; }

  const int mh = (lane >> 4) << 3;
  const int ncol = lane & 15;
  const int ntiles = blockIdx.x * 2 + 2;  // uniform across the block

  for (int t = 0; t < ntiles; ++t) {
    const int kb0 = t * 32;
    // ---- async-stage K and V 32x128 tiles once per block
#pragma unroll
    for (int j = 0; j < 4; ++j) {
      int e = tid * 8 + j * 1024;  // 0..4095, 8-half chunks
      int key = e >> 7, d = e & 127;
      copy16_g2l(Kp + (size_t)(kb0 + key) * HD + d, &sK[e]);
      copy16_g2l(Vp + (size_t)(kb0 + key) * HD + d, &sV[e]);
    }
    wait_async_lds();
    __syncthreads();

    // ---- scores: two 16-key C tiles, each 4 WMMAs over d (K tile from LDS)
    f32x8 s[2];
#pragma unroll
    for (int half = 0; half < 2; ++half) {
      f32x8 c;
#pragma unroll
      for (int r = 0; r < 8; ++r) c[r] = 0.0f;
#pragma unroll
      for (int kk = 0; kk < 4; ++kk) {
        bf16x16 bk = load_frag_b_t(sK + (size_t)(half * 16) * HD + kk * 32, HD);
        c = wmma_bf16(qfrag[kk], bk, c);
      }
#pragma unroll
      for (int r = 0; r < 8; ++r) {
        int qrow = q0 + r + mh;
        int kcol = kb0 + half * 16 + ncol;
        if (kcol > qrow) c[r] += NEG_INF_F;
      }
      s[half] = c;
    }
    // ---- online softmax (row reduce across 16-lane halves; wave32)
    float mnew[8], alpha[8];
#pragma unroll
    for (int r = 0; r < 8; ++r) {
      float v = fmaxf(s[0][r], s[1][r]);
      v = fmaxf(v, __shfl_xor(v, 1, 32));
      v = fmaxf(v, __shfl_xor(v, 2, 32));
      v = fmaxf(v, __shfl_xor(v, 4, 32));
      v = fmaxf(v, __shfl_xor(v, 8, 32));
      mnew[r] = fmaxf(mrow[r], v);
      alpha[r] = __expf(mrow[r] - mnew[r]);
      float p0 = __expf(s[0][r] - mnew[r]);
      float p1 = __expf(s[1][r] - mnew[r]);
      s[0][r] = p0;
      s[1][r] = p1;
      float rs = p0 + p1;
      rs += __shfl_xor(rs, 1, 32);
      rs += __shfl_xor(rs, 2, 32);
      rs += __shfl_xor(rs, 4, 32);
      rs += __shfl_xor(rs, 8, 32);
      lrow[r] = lrow[r] * alpha[r] + rs;
      mrow[r] = mnew[r];
    }
#pragma unroll
    for (int d = 0; d < 8; ++d)
#pragma unroll
      for (int r = 0; r < 8; ++r) o[d][r] *= alpha[r];
    // ---- P -> per-wave LDS (C layout -> A layout source)
    __bf16* P = sP[wave];
#pragma unroll
    for (int half = 0; half < 2; ++half)
#pragma unroll
      for (int r = 0; r < 8; ++r)
        P[(r + mh) * 32 + half * 16 + ncol] = to_bf16(s[half][r]);
    asm volatile("s_wait_dscnt 0" ::: "memory");  // in-wave LDS RAW handoff
    // ---- O += P x V (V read column-wise from shared row-major tile)
    bf16x16 pf = load_frag_a(P, 32);
#pragma unroll
    for (int dt = 0; dt < 8; ++dt) {
      bf16x16 vfr = load_frag_b_rm(sV + dt * 16, HD);
      o[dt] = wmma_bf16(pf, vfr, o[dt]);
    }
    __syncthreads();  // all waves done with sK/sV before next stage
  }

  // ---- normalize and write attn output (B, L, H*HD) as bf16
  __bf16* op = attnb + ((size_t)(b * L_SEQ + q0 + mh)) * DMODEL + h * HD + ncol;
#pragma unroll
  for (int dt = 0; dt < 8; ++dt)
#pragma unroll
    for (int r = 0; r < 8; ++r)
      op[(size_t)r * DMODEL + dt * 16] = to_bf16(o[dt][r] / lrow[r]);
}

// ---------------- host-side orchestration ----------------------------------
extern "C" void kernel_launch(void* const* d_in, const int* in_sizes, int n_in,
                              void* d_out, int out_size, void* d_ws, size_t ws_size,
                              hipStream_t stream) {
  (void)in_sizes; (void)n_in; (void)out_size; (void)ws_size;
  const float* x = (const float*)d_in[0];
  // d_in[1] = mask (unused; causal mask computed analytically)
  const float* wq = (const float*)d_in[2];
  const float* wk = (const float*)d_in[3];
  const float* wv = (const float*)d_in[4];
  const float* wo = (const float*)d_in[5];
  float* out = (float*)d_out;
  const size_t OUT_K = (size_t)BATCH * L_SEQ * DMODEL;  // 16777216
  const size_t OUT_V = 2 * OUT_K;

  char* ws = (char*)d_ws;
  size_t off = 0;
  auto alloc = [&](size_t bytes) -> void* {
    void* p = ws + off;
    off += (bytes + 255) & ~(size_t)255;
    return p;
  };
  const size_t ROWS = (size_t)BATCH * L_SEQ;  // 4096
  __bf16* xb   = (__bf16*)alloc(ROWS * DMODEL * 2);
  __bf16* wqbT = (__bf16*)alloc((size_t)DMODEL * DMODEL * 2);
  __bf16* wkbT = (__bf16*)alloc((size_t)DMODEL * NKV * HD * 2);
  __bf16* wvbT = (__bf16*)alloc((size_t)DMODEL * NKV * HD * 2);
  __bf16* wobT = (__bf16*)alloc((size_t)DMODEL * DMODEL * 2);
  float*  qf   = (float*)alloc(ROWS * DMODEL * 4);
  float*  kf   = (float*)alloc(ROWS * NKV * HD * 4);
  float*  vf   = (float*)alloc(ROWS * NKV * HD * 4);
  __bf16* qbh  = (__bf16*)alloc((size_t)BATCH * NH * L_SEQ * HD * 2);
  __bf16* kbh  = (__bf16*)alloc((size_t)BATCH * NKV * L_SEQ * HD * 2);
  __bf16* vbh  = (__bf16*)alloc((size_t)BATCH * NKV * L_SEQ * HD * 2);
  __bf16* attnb = (__bf16*)alloc(ROWS * DMODEL * 2);

  cvt_bf16<<<2048, 256, 0, stream>>>(x, (unsigned short*)xb, ROWS * DMODEL);
  cvt_t_bf16<12><<<2048, 256, 0, stream>>>(wq, (unsigned short*)wqbT, DMODEL);
  cvt_t_bf16<10><<<1024, 256, 0, stream>>>(wk, (unsigned short*)wkbT, DMODEL);
  cvt_t_bf16<10><<<1024, 256, 0, stream>>>(wv, (unsigned short*)wvbT, DMODEL);
  cvt_t_bf16<12><<<2048, 256, 0, stream>>>(wo, (unsigned short*)wobT, DMODEL);

  dim3 gq(DMODEL / GT_N, (int)(ROWS / GT_M));
  dim3 gkv((NKV * HD) / GT_N, (int)(ROWS / GT_M));
  gemm_bf16<DMODEL, DMODEL><<<gq, 256, 0, stream>>>(xb, wqbT, qf, (int)ROWS);
  gemm_bf16<NKV * HD, DMODEL><<<gkv, 256, 0, stream>>>(xb, wkbT, kf, (int)ROWS);
  gemm_bf16<NKV * HD, DMODEL><<<gkv, 256, 0, stream>>>(xb, wvbT, vf, (int)ROWS);

  rope_q_kernel<<<(BATCH * L_SEQ * NH * 64) / 256, 256, 0, stream>>>(qf, qbh);
  rope_k_kernel<<<(BATCH * L_SEQ * NKV * 64) / 256, 256, 0, stream>>>(kf, kbh, out + OUT_K);
  copy_v_kernel<<<(BATCH * L_SEQ * NKV * HD) / 256, 256, 0, stream>>>(vf, vbh, out + OUT_V);

  dim3 gf(L_SEQ / 64, NH, BATCH);
  flash_attn<<<gf, 128, 0, stream>>>(qbh, kbh, vbh, attnb);

  gemm_bf16<DMODEL, DMODEL><<<gq, 256, 0, stream>>>(attnb, wobT, out, (int)ROWS);
}
